// CausalAttention_38250978738352
// MI455X (gfx1250) — matrix operand order repaired and checked by reading, not compiled, not run on hardware
//
#include <hip/hip_runtime.h>
#include <hip/hip_bf16.h>

typedef _Float16 f16;
typedef __attribute__((ext_vector_type(8)))  _Float16 v8h;
typedef __attribute__((ext_vector_type(16))) _Float16 v16h;
typedef __attribute__((ext_vector_type(8)))  float    v8f;

typedef unsigned int u32x4 __attribute__((ext_vector_type(4)));
typedef int          i32x4 __attribute__((ext_vector_type(4)));
typedef int          i32x8 __attribute__((ext_vector_type(8)));

#define DIM    1024
#define HEADS  16
#define HD     64
#define BATCH  2
#define NSEQ   2048
#define MEMLEN 512
#define JSEQ   (MEMLEN + NSEQ)   // 2560

#if defined(__HIP_DEVICE_COMPILE__) && defined(__has_builtin)
#  if __has_builtin(__builtin_amdgcn_tensor_load_to_lds) && __has_builtin(__builtin_amdgcn_s_wait_tensorcnt)
#    define USE_TDM 1
#  endif
#endif
#ifndef USE_TDM
#  define USE_TDM 0
#endif

static __device__ inline v8f v8f_zero() {
    v8f z;
    #pragma unroll
    for (int i = 0; i < 8; ++i) z[i] = 0.0f;
    return z;
}

static __device__ inline v8f wmma16(v16h a, v16h b, v8f c) {
    // D = A(16x32 f16) * B(32x16 f16) + C(16x16 f32)
    return __builtin_amdgcn_wmma_f32_16x16x32_f16(false, a, false, b, (short)0, c, false, false);
}

static __device__ inline v8h ld8(const f16* p) { return *(const v8h*)p; }

static __device__ inline v16h cat8(v8h lo, v8h hi) {
    return __builtin_shufflevector(lo, hi, 0, 1, 2, 3, 4, 5, 6, 7,
                                            8, 9, 10, 11, 12, 13, 14, 15);
}

static __device__ inline v16h frag_from(const f16* p, const f16* p2) {
    return cat8(ld8(p), ld8(p2));
}

static __device__ inline v8h conv8(const float* p) {
    float4 a = ((const float4*)p)[0];
    float4 b = ((const float4*)p)[1];
    v8h o;
    o[0] = (f16)a.x; o[1] = (f16)a.y; o[2] = (f16)a.z; o[3] = (f16)a.w;
    o[4] = (f16)b.x; o[5] = (f16)b.y; o[6] = (f16)b.z; o[7] = (f16)b.w;
    return o;
}

#if USE_TDM
// Tensor Data Mover: DMA a 2D f16 tile (tile_w x tile_h, row stride row_stride
// elements) from global memory into LDS at byte offset lds_addr, inserting
// pad_amount(+1 coded) DWORDs of padding every pad_interval(coded) DWORDs so the
// LDS layout matches our padded tiles.  D# layout per CDNA5 ISA.
static __device__ inline void tdm_load_2d(unsigned lds_addr, const void* gptr,
                                          unsigned tile_w, unsigned tile_h,
                                          unsigned row_stride,
                                          unsigned tensor_w, unsigned tensor_h,
                                          unsigned pad_interval_code,
                                          unsigned pad_amount_code) {
    unsigned long long ga = (unsigned long long)gptr;
    u32x4 g0;
    g0[0] = 1u;                                     // count=1 (valid user descriptor)
    g0[1] = lds_addr;                               // lds_addr, bits 63:32
    g0[2] = (unsigned)(ga & 0xFFFFFFFFu);           // global_addr[31:0]  (bit 64)
    g0[3] = (unsigned)((ga >> 32) & 0x1FFFFFFu)     // global_addr[56:32]
          | (2u << 30);                             // type=2 ("image"), bits 127:126
    i32x8 g1;
    g1[0] = (int)((1u << 16)                        // data_size = 1 -> 2 bytes
          | (1u << 20)                              // pad_enable
          | (pad_interval_code << 22)               // pad_interval
          | (pad_amount_code << 25));               // pad_amount
    g1[1] = (int)((tensor_w & 0xFFFFu) << 16);      // tensor_dim0[15:0] @ bit 48
    g1[2] = (int)(((tensor_w >> 16) & 0xFFFFu)      // tensor_dim0[31:16]
          | ((tensor_h & 0xFFFFu) << 16));          // tensor_dim1[15:0] @ bit 80
    g1[3] = (int)(((tensor_h >> 16) & 0xFFFFu)      // tensor_dim1[31:16]
          | ((tile_w & 0xFFFFu) << 16));            // tile_dim0 @ bit 112
    g1[4] = (int)(tile_h & 0xFFFFu);                // tile_dim1 @ bit 128 (tile_dim2 = 0)
    g1[5] = (int)row_stride;                        // tensor_dim0_stride[31:0] @ bit 160
    g1[6] = 0;                                      // dim0_stride[47:32], dim1_stride lo
    g1[7] = 0;
    i32x4 z4 = {0, 0, 0, 0};
#if __clang_major__ >= 23
    i32x8 z8 = {0, 0, 0, 0, 0, 0, 0, 0};
    __builtin_amdgcn_tensor_load_to_lds(g0, g1, z4, z4, z8, 0);
#else
    __builtin_amdgcn_tensor_load_to_lds(g0, g1, z4, z4, 0);
#endif
}
#endif // USE_TDM

// ---------------------------------------------------------------------------
// vectorized f32 -> f16 (n8 = count/8)
// ---------------------------------------------------------------------------
__global__ void k_f32_to_f16(const float* __restrict__ src, f16* __restrict__ dst, int n8) {
    int i = blockIdx.x * blockDim.x + threadIdx.x;
    int stride = gridDim.x * blockDim.x;
    for (; i < n8; i += stride)
        ((v8h*)dst)[i] = conv8(src + (size_t)i * 8);
}

// xc[b][J][DIM] = concat(mem[b], x[b]) as f16, in 8-element chunks
__global__ void k_concat_f16(const float* __restrict__ mem, const float* __restrict__ x,
                             f16* __restrict__ xc) {
    const int CH = DIM / 8;                      // 128 chunks per row
    const int total8 = BATCH * JSEQ * CH;
    int i = blockIdx.x * blockDim.x + threadIdx.x;
    int stride = gridDim.x * blockDim.x;
    for (; i < total8; i += stride) {
        int b   = i / (JSEQ * CH);
        int rem = i - b * (JSEQ * CH);
        int row = rem / CH;
        int cc  = rem - row * CH;
        const float* src = (row < MEMLEN)
            ? mem + ((size_t)b * MEMLEN + row) * DIM + cc * 8
            : x + ((size_t)b * NSEQ + (row - MEMLEN)) * DIM + cc * 8;
        ((v8h*)xc)[i] = conv8(src);
    }
}

// ---------------------------------------------------------------------------
// Tiled WMMA GEMM: C(MxN) = A(MxK) @ B(KxN), f16 in, f16 or f32(+bias) out.
// Block = 128 threads (4 waves); block tile 128(M) x 64(N); wave tile 32x64;
// K-step 32 -> 8 WMMAs per wave per step.  A tile staged via TDM (DMA with
// LDS padding -> stride 40), B tile staged manually (needs transpose).
// ---------------------------------------------------------------------------
template <bool F32OUT>
__global__ __launch_bounds__(128) void k_gemm(const f16* __restrict__ A,
                                              const f16* __restrict__ B,
                                              void* __restrict__ Cv,
                                              const float* __restrict__ bias,
                                              int M, int N, int K) {
    __shared__ __align__(16) f16 At[128 * 40];   // A tile row-major, stride 40
    __shared__ __align__(16) f16 Bt[64 * 40];    // B tile transposed: Bt[n][k]

    const int tid  = threadIdx.x;
    const int lane = tid & 31;
    const int wid  = tid >> 5;
    const int half = lane >> 4;
    const int r    = lane & 15;

    const int row0 = blockIdx.y * 128;
    const int col0 = blockIdx.x * 64;

    v8f acc[2][4];
    #pragma unroll
    for (int mi = 0; mi < 2; ++mi)
        #pragma unroll
        for (int nt = 0; nt < 4; ++nt) acc[mi][nt] = v8f_zero();

    for (int k0 = 0; k0 < K; k0 += 32) {
#if USE_TDM
        // ---- stage A tile (128x32) via Tensor Data Mover ----
        // row = 32 f16 = 16 DW, pad 4 DW -> LDS stride 20 DW = 40 f16.
        if (wid == 0) {
            tdm_load_2d((unsigned)(size_t)(void*)At,
                        A + (size_t)row0 * K + k0,
                        /*tile_w=*/32, /*tile_h=*/128,
                        /*row_stride=*/(unsigned)K,
                        /*tensor_w=*/(unsigned)K, /*tensor_h=*/1u << 20,
                        /*pad_interval (16 DW)=*/3, /*pad_amount (4 DW)=*/3);
            __builtin_amdgcn_s_wait_tensorcnt(0);
        }
#else
        #pragma unroll
        for (int c2 = 0; c2 < 4; ++c2) {
            int ci = tid + c2 * 128;
            int ar = ci >> 2;
            int ac = (ci & 3) * 8;
            *(v8h*)(At + ar * 40 + ac) = ld8(A + (size_t)(row0 + ar) * K + k0 + ac);
        }
#endif
        // ---- stage B tile (32x64) transposed: 256 chunks, 2 per thread ----
        #pragma unroll
        for (int c2 = 0; c2 < 2; ++c2) {
            int ci = tid + c2 * 128;
            int kr = ci >> 3;
            int nc = (ci & 7) * 8;
            v8h v = ld8(B + (size_t)(k0 + kr) * N + col0 + nc);
            #pragma unroll
            for (int j = 0; j < 8; ++j) Bt[(nc + j) * 40 + kr] = v[j];
        }
        // prefetch next K-step B tile into cache while we compute
        if (k0 + 32 < K) {
            int kr = tid >> 3, nc = (tid & 7) * 8;
            __builtin_prefetch(B + (size_t)(k0 + 32 + kr) * N + col0 + nc, 0, 3);
        }
        __syncthreads();

        // ---- wave tile: rows wid*32 .. wid*32+31 ----
        #pragma unroll
        for (int mi = 0; mi < 2; ++mi) {
            const f16* arow = At + (wid * 32 + mi * 16 + r) * 40 + half * 8;
            v16h a = frag_from(arow, arow + 16);
            #pragma unroll
            for (int nt = 0; nt < 4; ++nt) {
                const f16* brow = Bt + (nt * 16 + r) * 40 + half * 8;
                v16h bf = frag_from(brow, brow + 16);
                acc[mi][nt] = wmma16(a, bf, acc[mi][nt]);
            }
        }
        __syncthreads();
    }

    // epilogue: C layout — VGPR v, lanes 0-15 -> M=v, lanes 16-31 -> M=8+v, N=lane&15
    #pragma unroll
    for (int mi = 0; mi < 2; ++mi) {
        #pragma unroll
        for (int nt = 0; nt < 4; ++nt) {
            #pragma unroll
            for (int vr = 0; vr < 8; ++vr) {
                int rowm = row0 + wid * 32 + mi * 16 + half * 8 + vr;
                int coln = col0 + nt * 16 + r;
                float val = acc[mi][nt][vr];
                if (F32OUT) {
                    float bb = bias ? bias[coln] : 0.0f;
                    ((float*)Cv)[(size_t)rowm * N + coln] = val + bb;
                } else {
                    ((f16*)Cv)[(size_t)rowm * N + coln] = (f16)val;
                }
            }
        }
    }
}

// ---------------------------------------------------------------------------
// Flash attention with memory prefix.
// Q:  [b*N, DIM] f16 ; KV: [b*J, 2*DIM] f16 (K cols 0..1023, V cols 1024..2047)
// O:  [b*N, DIM] f16
// Block = 128 threads (4 waves) owning 64 query rows of one (b,h);
// K tile staged via TDM (padded stride 72), V tile staged manually (transposed).
// ---------------------------------------------------------------------------
__global__ __launch_bounds__(128) void k_attn(const f16* __restrict__ Q,
                                              const f16* __restrict__ KV,
                                              f16* __restrict__ O) {
    const int tid  = threadIdx.x;
    const int lane = tid & 31;
    const int wid  = tid >> 5;
    const int half = lane >> 4;
    const int r    = lane & 15;

    const int hh = blockIdx.y;
    const int bb = blockIdx.z;
    const int i0_blk = blockIdx.x * 64;
    const int i0 = i0_blk + wid * 16;
    const float scale = 0.125f;  // 64^-0.5

    __shared__ __align__(16) f16 Kt[32 * 72];        // K tile row-major [key][d], stride 72
    __shared__ __align__(16) f16 Vt[64 * 40];        // V tile transposed [d][key], stride 40
    __shared__ __align__(16) f16 ldsP[4][16 * 40];   // per-wave P scratch
    f16* myP = ldsP[wid];

    // Q fragments, loaded once (head-dim chunks 0..31 and 32..63)
    const f16* qrow = Q + ((size_t)bb * NSEQ + i0 + r) * DIM + hh * HD;
    v16h qa[2];
    #pragma unroll
    for (int kk = 0; kk < 2; ++kk) {
        const f16* p = qrow + kk * 32 + half * 8;
        qa[kk] = frag_from(p, p + 16);
    }

    v8f acc[4];
    #pragma unroll
    for (int t = 0; t < 4; ++t) acc[t] = v8f_zero();
    float mrow[8], lrow[8];
    #pragma unroll
    for (int vr = 0; vr < 8; ++vr) { mrow[vr] = -1e30f; lrow[vr] = 0.0f; }

    const size_t kvbase = (size_t)bb * JSEQ * (2 * DIM) + hh * HD;
    const int jhi_wave = i0 + MEMLEN + 16;        // this wave's exclusive key bound
    const int jhi_blk  = i0_blk + MEMLEN + 64;    // block-uniform bound (<= JSEQ)

    for (int j0 = 0; j0 < jhi_blk; j0 += 32) {
#if USE_TDM
        // ---- K tile (32 keys x 64 d) via TDM: row 32 DW + 4 DW pad = stride 72 f16
        if (wid == 0) {
            tdm_load_2d((unsigned)(size_t)(void*)Kt,
                        KV + kvbase + (size_t)j0 * (2 * DIM),
                        /*tile_w=*/64, /*tile_h=*/32,
                        /*row_stride=*/2 * DIM,
                        /*tensor_w=*/2 * DIM, /*tensor_h=*/1u << 20,
                        /*pad_interval (32 DW)=*/4, /*pad_amount (4 DW)=*/3);
            __builtin_amdgcn_s_wait_tensorcnt(0);
        }
        // ---- V tile staged manually (needs transpose) ----
        #pragma unroll
        for (int c2 = 0; c2 < 2; ++c2) {
            int ci = tid + c2 * 128;
            int kr = ci >> 3;
            int nc = (ci & 7) * 8;
            v8h v = ld8(KV + kvbase + DIM + (size_t)(j0 + kr) * (2 * DIM) + nc);
            #pragma unroll
            for (int j = 0; j < 8; ++j) Vt[(nc + j) * 40 + kr] = v[j];
        }
#else
        #pragma unroll
        for (int c2 = 0; c2 < 2; ++c2) {
            int ci = tid + c2 * 128;
            int kr = ci >> 3;            // key row in tile (0..31)
            int nc = (ci & 7) * 8;       // d offset (0..56)
            const f16* src = KV + kvbase + (size_t)(j0 + kr) * (2 * DIM) + nc;
            *(v8h*)(Kt + kr * 72 + nc) = ld8(src);
            v8h v = ld8(src + DIM);
            #pragma unroll
            for (int j = 0; j < 8; ++j) Vt[(nc + j) * 40 + kr] = v[j];
        }
#endif
        if (j0 + 32 < jhi_blk) {
            int kr = tid >> 3, nc = (tid & 7) * 8;
            __builtin_prefetch(KV + kvbase + (size_t)(j0 + 32 + kr) * (2 * DIM) + nc, 0, 3);
        }
        __syncthreads();

        if (j0 < jhi_wave) {
            // ---- S = (q @ k^T) for keys [j0, j0+32) ----
            v8f S0 = v8f_zero(), S1 = v8f_zero();
            #pragma unroll
            for (int kk = 0; kk < 2; ++kk) {
                const f16* kp0 = Kt + r * 72 + kk * 32 + half * 8;
                const f16* kp1 = Kt + (16 + r) * 72 + kk * 32 + half * 8;
                v16h b0 = frag_from(kp0, kp0 + 16);
                v16h b1 = frag_from(kp1, kp1 + 16);
                S0 = wmma16(qa[kk], b0, S0);
                S1 = wmma16(qa[kk], b1, S1);
            }

            const bool tail = (j0 + 31) > (i0 + MEMLEN);

            // ---- online softmax (rows spread across 16-lane groups) ----
            #pragma unroll
            for (int vr = 0; vr < 8; ++vr) {
                float s0 = S0[vr] * scale;
                float s1 = S1[vr] * scale;
                if (tail) {
                    int rowi = i0 + half * 8 + vr;
                    if (j0 + r      > rowi + MEMLEN) s0 = -1e30f;
                    if (j0 + 16 + r > rowi + MEMLEN) s1 = -1e30f;
                }
                float cand = fmaxf(s0, s1);
                #pragma unroll
                for (int off = 8; off >= 1; off >>= 1)
                    cand = fmaxf(cand, __shfl_xor(cand, off, 16));
                float mnew  = fmaxf(mrow[vr], cand);
                float alpha = __expf(mrow[vr] - mnew);
                float p0 = __expf(s0 - mnew);
                float p1 = __expf(s1 - mnew);
                float rs = p0 + p1;
                #pragma unroll
                for (int off = 8; off >= 1; off >>= 1)
                    rs += __shfl_xor(rs, off, 16);
                mrow[vr] = mnew;
                lrow[vr] = lrow[vr] * alpha + rs;
                #pragma unroll
                for (int nt = 0; nt < 4; ++nt) acc[nt][vr] = acc[nt][vr] * alpha;
                myP[(half * 8 + vr) * 40 + r]      = (f16)p0;
                myP[(half * 8 + vr) * 40 + 16 + r] = (f16)p1;
            }

            // re-load P as an A-fragment (in-wave LDS ops stay ordered)
            const f16* prow = myP + r * 40 + half * 8;
            v16h pa = frag_from(prow, prow + 16);

            // ---- acc += P(16x32) @ V(32x64), V fragments contiguous in Vt ----
            #pragma unroll
            for (int nt = 0; nt < 4; ++nt) {
                const f16* vp = Vt + (nt * 16 + r) * 40 + half * 8;
                v16h vb = frag_from(vp, vp + 16);
                acc[nt] = wmma16(pa, vb, acc[nt]);
            }
        }
        __syncthreads();
    }

    // ---- normalize and store ----
    #pragma unroll
    for (int nt = 0; nt < 4; ++nt) {
        #pragma unroll
        for (int vr = 0; vr < 8; ++vr) {
            int rowm = i0 + half * 8 + vr;
            float v = acc[nt][vr] / lrow[vr];
            O[((size_t)bb * NSEQ + rowm) * DIM + hh * HD + nt * 16 + r] = (f16)v;
        }
    }
}

// ---------------------------------------------------------------------------
extern "C" void kernel_launch(void* const* d_in, const int* in_sizes, int n_in,
                              void* d_out, int out_size, void* d_ws, size_t ws_size,
                              hipStream_t stream) {
    const float* x    = (const float*)d_in[0];
    const float* mem  = (const float*)d_in[1];
    const float* Wq   = (const float*)d_in[2];
    const float* Wkv  = (const float*)d_in[3];
    const float* Wout = (const float*)d_in[4];
    const float* bout = (const float*)d_in[5];
    float* out = (float*)d_out;

    char* ws = (char*)d_ws;
    size_t off = 0;
    auto alloc = [&](size_t bytes) -> char* {
        char* p = ws + off;
        off += (bytes + 255) & ~(size_t)255;
        return p;
    };
    f16* xh    = (f16*)alloc((size_t)BATCH * NSEQ * DIM * 2);
    f16* xch   = (f16*)alloc((size_t)BATCH * JSEQ * DIM * 2);
    f16* Wqh   = (f16*)alloc((size_t)DIM * DIM * 2);
    f16* Wkvh  = (f16*)alloc((size_t)DIM * 2 * DIM * 2);
    f16* Wouth = (f16*)alloc((size_t)DIM * DIM * 2);
    f16* qh    = (f16*)alloc((size_t)BATCH * NSEQ * DIM * 2);
    f16* kvh   = (f16*)alloc((size_t)BATCH * JSEQ * 2 * DIM * 2);
    f16* aoh   = (f16*)alloc((size_t)BATCH * NSEQ * DIM * 2);
    (void)ws_size; (void)in_sizes; (void)n_in; (void)out_size;

    k_f32_to_f16<<<1024, 256, 0, stream>>>(x, xh, BATCH * NSEQ * DIM / 8);
    k_concat_f16<<<1024, 256, 0, stream>>>(mem, x, xch);
    k_f32_to_f16<<<256, 256, 0, stream>>>(Wq, Wqh, DIM * DIM / 8);
    k_f32_to_f16<<<512, 256, 0, stream>>>(Wkv, Wkvh, DIM * 2 * DIM / 8);
    k_f32_to_f16<<<256, 256, 0, stream>>>(Wout, Wouth, DIM * DIM / 8);

    // q = x @ Wq                      [4096 x 1024]
    k_gemm<false><<<dim3(DIM / 64, (BATCH * NSEQ) / 128), 128, 0, stream>>>(
        xh, Wqh, qh, nullptr, BATCH * NSEQ, DIM, DIM);
    // kv = concat(mem,x) @ Wkv        [5120 x 2048]
    k_gemm<false><<<dim3((2 * DIM) / 64, (BATCH * JSEQ) / 128), 128, 0, stream>>>(
        xch, Wkvh, kvh, nullptr, BATCH * JSEQ, 2 * DIM, DIM);
    // flash attention
    k_attn<<<dim3(NSEQ / 64, HEADS, BATCH), 128, 0, stream>>>(qh, kvh, aoh);
    // out = attn_out @ Wout + bout    (f32 output)
    k_gemm<true><<<dim3(DIM / 64, (BATCH * NSEQ) / 128), 128, 0, stream>>>(
        aoh, Wouth, out, bout, BATCH * NSEQ, DIM, DIM);
}
